// Backbone3D_11493332484202
// MI455X (gfx1250) — matrix operand order repaired and checked by reading, not compiled, not run on hardware
//
#include <hip/hip_runtime.h>
#include <hip/hip_bf16.h>

// ============================================================================
// MI455X (gfx1250, wave32) sparse 3D U-Net.
//
// Activations: f16 channels-last [B][D][H][W][Cs], Cs = align16(C), pad
// channels zeroed. Implicit-GEMM K-order k = tap*Cs + ci so each 8-element
// K-run of a WMMA A-fragment is 16 contiguous bytes -> global_load_b128,
// no LDS, no barriers. Weights packed column-major bmatT[Npad][Kalloc] f16
// -> each lane's B-fragment is two 8-half runs 16 halves apart (2x b128, L2 resident).
// Each wave computes two 16x16 D tiles sharing the B fragment (2 WMMAs per
// K-chunk). Transposed convs = 8 phase-wise 1x1 GEMMs.
// All integer decode uses 32-bit magic-multiply division (host-precomputed
// m = ceil(2^32/d), exact since n*d < 2^32 for every divide in this net);
// batch rides on blockIdx.z so no 64-bit div/mod anywhere.
// ============================================================================

typedef __attribute__((ext_vector_type(16))) _Float16 v16h;
typedef __attribute__((ext_vector_type(8)))  _Float16 v8h;
typedef __attribute__((ext_vector_type(8)))  float    v8f;
typedef _Float16 half_t;

#define WPB 8                  // waves per block
#define CONV_BLOCK (WPB * 32)
#define SHUF16(lo, hi) \
  __builtin_shufflevector(lo, hi, 0, 1, 2, 3, 4, 5, 6, 7, 8, 9, 10, 11, 12, 13, 14, 15)

__device__ __forceinline__ int fdiv(int n, unsigned m) {
  // m = 0 encodes divisor 1
  return m ? (int)__umulhi((unsigned)n, m) : n;
}

struct ConvDesc {
  const half_t* x;      // input, channels-last f16, Cs channels/voxel
  const half_t* bmatT;  // packed weights [Npad][Kalloc] f16
  half_t*       y;      // output, channels-last f16, OCs channels/voxel
  int Cs;      unsigned mCs;
  int TAPS, KHW, KW;  unsigned mKHW, mKW;
  int ID, IH, IW;      // input spatial dims
  int OH, OW;          // output H,W (fine grid for mode==1)
  int stride, pad;
  int CV, CHW, CW;  unsigned mCHW, mCW;   // compute grid (out / coarse)
  int Kpad, Kalloc, Npad;
  int mode;            // 0 = conv, 1 = transposed-conv phase (1x1)
  int pd, ph, pw;      // phase for mode 1
  int OCs, oc0;        // output stored channels and channel offset
  int IVC, OVC;        // per-image strides: ID*IH*IW*Cs, OV*OCs
};

__global__ __launch_bounds__(CONV_BLOCK)
void conv_wmma_kernel(ConvDesc d) {
  const int lane = threadIdx.x & 31;
  const int wv   = threadIdx.x >> 5;
  const int mbase = (blockIdx.x * WPB + wv) * 32;  // 2 M-tiles of 16 voxels

  const half_t* xn = d.x + (long long)blockIdx.z * d.IVC;
  half_t*       yn = d.y + (long long)blockIdx.z * d.OVC;

  const int mrowA  = lane & 15;
  const int khalfA = (lane >> 4) * 8;    // A-fragment K offset (0 or 8)
  const int koffB  = (lane >> 4) * 8;    // B-fragment K offset (0 or 8)
  const int ncol   = blockIdx.y * 16 + (lane & 15);

  // ---- hoisted voxel decode: one A-row voxel per lane per tile ----
  int id0[2], ih0[2], iw0[2];
  bool av[2];
#pragma unroll
  for (int t = 0; t < 2; ++t) {
    const int mg = mbase + t * 16 + mrowA;
    av[t] = (mg < d.CV);
    const int vv = av[t] ? mg : 0;
    const int cd = fdiv(vv, d.mCHW);
    const int r2 = vv - cd * d.CHW;
    const int ch = fdiv(r2, d.mCW);
    const int cw = r2 - ch * d.CW;
    id0[t] = cd * d.stride - d.pad;
    ih0[t] = ch * d.stride - d.pad;
    iw0[t] = cw * d.stride - d.pad;
  }

  const v8h* bcol = (const v8h*)(d.bmatT + (size_t)ncol * d.Kalloc);
  __builtin_prefetch((const void*)bcol, 0, 1);  // -> global_prefetch_b8

  v8f acc[2];
  acc[0] = (v8f){0.f, 0.f, 0.f, 0.f, 0.f, 0.f, 0.f, 0.f};
  acc[1] = (v8f){0.f, 0.f, 0.f, 0.f, 0.f, 0.f, 0.f, 0.f};
  const v8h zero8 = {0, 0, 0, 0, 0, 0, 0, 0};

  for (int k0 = 0; k0 < d.Kpad; k0 += 32) {
    // decode the two 8-element K-runs this lane owns (run1 = run0 + 16)
    const int kr0 = k0 + khalfA;
    const int kr1 = kr0 + 16;
    const int tap0 = fdiv(kr0, d.mCs), ci0 = kr0 - tap0 * d.Cs;
    const int tap1 = fdiv(kr1, d.mCs), ci1 = kr1 - tap1 * d.Cs;
    const int kd0 = fdiv(tap0, d.mKHW), kq0 = tap0 - kd0 * d.KHW;
    const int kh0 = fdiv(kq0, d.mKW),   kw0 = kq0 - kh0 * d.KW;
    const int kd1 = fdiv(tap1, d.mKHW), kq1 = tap1 - kd1 * d.KHW;
    const int kh1 = fdiv(kq1, d.mKW),   kw1 = kq1 - kh1 * d.KW;
    const bool tv0 = (tap0 < d.TAPS);
    const bool tv1 = (tap1 < d.TAPS);

    v16h a[2];
#pragma unroll
    for (int t = 0; t < 2; ++t) {
      v8h r0 = zero8, r1 = zero8;
      if (av[t] && tv0) {
        const int id = id0[t] + kd0, ih = ih0[t] + kh0, iw = iw0[t] + kw0;
        if ((unsigned)id < (unsigned)d.ID && (unsigned)ih < (unsigned)d.IH &&
            (unsigned)iw < (unsigned)d.IW) {
          r0 = *(const v8h*)(xn + ((id * d.IH + ih) * d.IW + iw) * d.Cs + ci0);
        }
      }
      if (av[t] && tv1) {
        const int id = id0[t] + kd1, ih = ih0[t] + kh1, iw = iw0[t] + kw1;
        if ((unsigned)id < (unsigned)d.ID && (unsigned)ih < (unsigned)d.IH &&
            (unsigned)iw < (unsigned)d.IW) {
          r1 = *(const v8h*)(xn + ((id * d.IH + ih) * d.IW + iw) * d.Cs + ci1);
        }
      }
      a[t] = SHUF16(r0, r1);
    }

    const v8h bl = bcol[(k0 + koffB) >> 3];
    const v8h bh = bcol[((k0 + koffB) >> 3) + 2];
    const v16h b = SHUF16(bl, bh);

    acc[0] = __builtin_amdgcn_wmma_f32_16x16x32_f16(false, a[0], false, b,
                                                    (short)0, acc[0], false, false);
    acc[1] = __builtin_amdgcn_wmma_f32_16x16x32_f16(false, a[1], false, b,
                                                    (short)0, acc[1], false, false);
  }

  // ---- epilogue: scatter D per 32-bit C/D 16x16 layout (channels-last) ----
  if (d.oc0 + ncol < d.OCs) {
#pragma unroll
    for (int t = 0; t < 2; ++t) {
#pragma unroll
      for (int r = 0; r < 8; ++r) {
        const int M  = t * 16 + r + ((lane >> 4) << 3);
        const int mg = mbase + M;
        if (mg < d.CV) {
          int od = fdiv(mg, d.mCHW);
          int r2 = mg - od * d.CHW;
          int oh = fdiv(r2, d.mCW);
          int ow = r2 - oh * d.CW;
          if (d.mode) { od = 2 * od + d.pd; oh = 2 * oh + d.ph; ow = 2 * ow + d.pw; }
          const int vox = (od * d.OH + oh) * d.OW + ow;
          yn[vox * d.OCs + d.oc0 + ncol] = (half_t)acc[t][r];
        }
      }
    }
  }
}

// ---- pack conv weights (OIDHW f32) into column-major f16 [Npad][Kalloc] ----
__global__ void pack_w_kernel(const float* w, half_t* bm, int Ci, int Co,
                              int Cs, unsigned mCs, int TAPS, int Kalloc,
                              unsigned mKalloc, int Kpad, int tot) {
  int i = blockIdx.x * blockDim.x + threadIdx.x;
  if (i >= tot) return;
  int n = fdiv(i, mKalloc), k = i - n * Kalloc;
  float v = 0.f;
  if (n < Co && k < Kpad) {
    int tap = fdiv(k, mCs), ci = k - tap * Cs;
    if (ci < Ci) v = w[(n * Ci + ci) * TAPS + tap];
  }
  bm[i] = (half_t)v;
}

// ---- pack one phase of a 2x2x2 transposed conv: k = ci ----
__global__ void pack_wt_kernel(const float* w, half_t* bm, int Ci, int Co,
                               int tap, int Kalloc, unsigned mKalloc, int tot) {
  int i = blockIdx.x * blockDim.x + threadIdx.x;
  if (i >= tot) return;
  int n = fdiv(i, mKalloc), k = i - n * Kalloc;
  float v = 0.f;
  if (n < Co && k < Ci) v = w[(n * Ci + k) * 8 + tap];
  bm[i] = (half_t)v;
}

// ---- convert input f32 NCDHW -> f16 channels-last [B][V][16] (C=2) ----
__global__ void convert_x_kernel(const float* x, half_t* y, int V, int total) {
  int i = blockIdx.x * blockDim.x + threadIdx.x;
  if (i >= total) return;
  const int n = blockIdx.y;
  int c = i & 15, vox = i >> 4;
  float v = (c < 2) ? x[((long long)n * 2 + c) * V + vox] : 0.f;
  y[(long long)n * total + i] = (half_t)v;
}

// ---- masked BN statistics (channels-last f16, f32 accumulate) ----
__global__ void bn_stats_kernel(const half_t* x, const unsigned char* mask,
                                float* stats, int Cs, int BV) {
  const int c = blockIdx.x;
  int idx = blockIdx.y * blockDim.x + threadIdx.x;
  const int step = gridDim.y * blockDim.x;
  float s = 0.f, ss = 0.f, cnt = 0.f;
  for (; idx < BV; idx += step) {
    float m  = mask[idx] ? 1.f : 0.f;
    float xv = (float)x[(long long)idx * Cs + c] * m;
    s += xv; ss += xv * xv; cnt += m;
  }
  __shared__ float sh[3][256];
  int t = threadIdx.x;
  sh[0][t] = s; sh[1][t] = ss; sh[2][t] = cnt;
  __syncthreads();
  for (int o = 128; o > 0; o >>= 1) {
    if (t < o) {
      sh[0][t] += sh[0][t + o];
      sh[1][t] += sh[1][t + o];
      sh[2][t] += sh[2][t + o];
    }
    __syncthreads();
  }
  if (t == 0) {
    atomicAdd(&stats[3 * c + 0], sh[0][0]);
    atomicAdd(&stats[3 * c + 1], sh[1][0]);
    atomicAdd(&stats[3 * c + 2], sh[2][0]);
  }
}

__global__ void bn_finalize_kernel(const float* stats, const float* g,
                                   const float* b, float* scsh, int C) {
  int c = blockIdx.x * blockDim.x + threadIdx.x;
  if (c >= C) return;
  float n    = fmaxf(stats[3 * c + 2], 1.f);
  float mean = stats[3 * c + 0] / n;
  float var  = stats[3 * c + 1] / n - mean * mean;
  float sc   = g[c] * rsqrtf(var + 1e-5f);
  scsh[2 * c + 0] = sc;
  scsh[2 * c + 1] = b[c] - mean * sc;
}

// ---- fused BN apply: y = relu?(((x*sc+sh)*mask)[+res]) ; scatter to concat
__global__ void bn_apply_kernel(const half_t* x, const float* scsh,
                                const unsigned char* mask, const half_t* res,
                                half_t* y, int C, int CsIn, unsigned mCsIn,
                                int CsRes, int CsTot, int oc0, int V,
                                int relu, int total /* V*CsIn */) {
  int i = blockIdx.x * blockDim.x + threadIdx.x;
  if (i >= total) return;
  const int n = blockIdx.y;
  int vox = fdiv(i, mCsIn);
  int c   = i - vox * CsIn;
  float val = 0.f;
  if (c < C) {
    float m = mask[(long long)n * V + vox] ? 1.f : 0.f;
    val = ((float)x[(long long)n * total + i] * scsh[2 * c] + scsh[2 * c + 1]) * m;
    if (res)  val += (float)res[((long long)n * V + vox) * CsRes + c];
    if (relu) val = fmaxf(val, 0.f);
  }
  y[((long long)n * V + vox) * CsTot + oc0 + c] = (half_t)val;
}

// ---- copy channel slice into a concat buffer (channels-last) ----
__global__ void copy_chan_kernel(const half_t* src, half_t* dst, int C,
                                 unsigned mC, int CsS, int CsT, int oc0,
                                 int V, int total /* V*C */) {
  int i = blockIdx.x * blockDim.x + threadIdx.x;
  if (i >= total) return;
  const int n = blockIdx.y;
  int vox = fdiv(i, mC);
  int c   = i - vox * C;
  dst[((long long)n * V + vox) * CsT + oc0 + c] =
      src[((long long)n * V + vox) * CsS + c];
}

// ---- final 1x1 conv + bias + mask + transpose (0,1,4,3,2) ----
// grid: (ceil(H*D/256), W, B*2)
__global__ void final_conv_kernel(const half_t* x, const float* wf,
                                  const float* bf, const unsigned char* m,
                                  float* out, int D, unsigned mD, int H,
                                  int W, int HD) {
  int i = blockIdx.x * blockDim.x + threadIdx.x;
  if (i >= HD) return;
  const int zz = blockIdx.y;
  const int nc = blockIdx.z;
  const int c = nc & 1, n = nc >> 1;
  int yy = fdiv(i, mD);
  int xx = i - yy * D;
  const int V = HD * W;
  const int vox = (xx * H + yy) * W + zz;
  const half_t* xb = x + (long long)n * V * 16;
  float mk = m[(long long)n * V + vox] ? 1.f : 0.f;
  float s = bf[c];
#pragma unroll
  for (int ci = 0; ci < 8; ++ci)
    s += (float)xb[vox * 16 + ci] * wf[c * 8 + ci];
  out[((long long)nc * W + zz) * HD + (long long)yy * D + xx] = s * mk;
}

// ============================================================================
// Host orchestration: bump allocator over d_ws; stream is serial so transient
// regions are safely reset between stages.
// ============================================================================

namespace {

struct Alloc { char* base; size_t off; size_t cap; };

inline void* amalloc(Alloc& a, size_t bytes) {
  size_t o = (a.off + 255) & ~(size_t)255;
  a.off = o + bytes;
  return a.base + o;
}
inline int align16i(int x) { return (x + 15) & ~15; }
inline int align32i(int x) { return (x + 31) & ~31; }
inline unsigned mkmag(unsigned d) {
  return d <= 1 ? 0u : (unsigned)((0x100000000ULL + d - 1) / d);
}

struct BlkP { const float *w1,*g1,*b1,*w2,*g2,*b2,*wd,*gd,*bd; };

void launch_conv(const ConvDesc& d, int B, hipStream_t s) {
  const int gx = (d.CV + 32 * WPB - 1) / (32 * WPB);
  conv_wmma_kernel<<<dim3(gx, d.Npad / 16, B), CONV_BLOCK, 0, s>>>(d);
}

void fill_common(ConvDesc& d, int Cs, int K, int CD, int CH, int CW) {
  d.Cs = Cs;   d.mCs  = mkmag(Cs);
  d.KHW = K * K; d.KW = K;
  d.mKHW = mkmag(K * K); d.mKW = mkmag(K);
  d.CV = CD * CH * CW; d.CHW = CH * CW; d.CW = CW;
  d.mCHW = mkmag(CH * CW); d.mCW = mkmag(CW);
}

void run_conv(const half_t* x, int Cs, int Ci, const float* w, half_t* yraw,
              int B, int Co, int ID, int IH, int IW, int K, int stride,
              int pad, int OD, int OH, int OW, Alloc& A, hipStream_t s) {
  const int TAPS = K * K * K;
  const int Kpad = TAPS * Cs;
  const int Kalloc = align32i(Kpad);
  const int Npad = align16i(Co);
  half_t* bm = (half_t*)amalloc(A, (size_t)Npad * Kalloc * sizeof(half_t));
  const int tot = Npad * Kalloc;
  pack_w_kernel<<<dim3((tot + 255) / 256), 256, 0, s>>>(
      w, bm, Ci, Co, Cs, mkmag(Cs), TAPS, Kalloc, mkmag(Kalloc), Kpad, tot);
  ConvDesc d{};
  d.x = x; d.bmatT = bm; d.y = yraw;
  fill_common(d, Cs, K, OD, OH, OW);
  d.TAPS = TAPS;
  d.ID = ID; d.IH = IH; d.IW = IW; d.OH = OH; d.OW = OW;
  d.stride = stride; d.pad = pad;
  d.Kpad = Kpad; d.Kalloc = Kalloc; d.Npad = Npad;
  d.mode = 0; d.pd = d.ph = d.pw = 0;
  d.OCs = Npad; d.oc0 = 0;
  d.IVC = ID * IH * IW * Cs;
  d.OVC = OD * OH * OW * Npad;
  launch_conv(d, B, s);
}

void run_convt(const half_t* x, int Cs, int Ci, const float* w, half_t* yraw,
               int B, int Co, int ID, int IH, int IW, Alloc& A, hipStream_t s) {
  const int Kpad = Cs;
  const int Kalloc = align32i(Kpad);
  const int Npad = align16i(Co);
  const int tot = Npad * Kalloc;
  for (int p = 0; p < 8; ++p) {
    half_t* bm = (half_t*)amalloc(A, (size_t)tot * sizeof(half_t));
    pack_wt_kernel<<<dim3((tot + 255) / 256), 256, 0, s>>>(
        w, bm, Ci, Co, p, Kalloc, mkmag(Kalloc), tot);
    ConvDesc d{};
    d.x = x; d.bmatT = bm; d.y = yraw;
    fill_common(d, Cs, 1, ID, IH, IW);   // compute grid = coarse grid
    d.TAPS = 1;
    d.ID = ID; d.IH = IH; d.IW = IW;
    d.OH = 2 * IH; d.OW = 2 * IW;
    d.stride = 1; d.pad = 0;
    d.Kpad = Kpad; d.Kalloc = Kalloc; d.Npad = Npad;
    d.mode = 1; d.pd = (p >> 2) & 1; d.ph = (p >> 1) & 1; d.pw = p & 1;
    d.OCs = Npad; d.oc0 = 0;
    d.IVC = ID * IH * IW * Cs;
    d.OVC = (8 * ID * IH * IW) * Npad;
    launch_conv(d, B, s);
  }
}

void run_bn(const half_t* raw, int CsIn, int C, const unsigned char* mask,
            const float* g, const float* bb, const half_t* res, int CsRes,
            half_t* y, int CsTot, int oc0, bool relu, int B, int V,
            Alloc& A, hipStream_t s) {
  float* stats = (float*)amalloc(A, (size_t)3 * C * sizeof(float));
  float* scsh  = (float*)amalloc(A, (size_t)2 * C * sizeof(float));
  (void)hipMemsetAsync(stats, 0, (size_t)3 * C * sizeof(float), s);
  bn_stats_kernel<<<dim3(C, 32), 256, 0, s>>>(raw, mask, stats, CsIn, B * V);
  bn_finalize_kernel<<<1, 128, 0, s>>>(stats, g, bb, scsh, C);
  const int total = V * CsIn;
  bn_apply_kernel<<<dim3((total + 255) / 256, B), 256, 0, s>>>(
      raw, scsh, mask, res, y, C, CsIn, mkmag(CsIn), CsRes, CsTot, oc0, V,
      relu ? 1 : 0, total);
}

void run_conv_bn(const half_t* xin, int CsIn, int Ci, const float* w,
                 const float* g, const float* b, const unsigned char* mask,
                 half_t* y, int B, int Co, int ID, int IH, int IW, int K,
                 int stride, int pad, int OD, int OH, int OW, Alloc& A,
                 hipStream_t s) {
  const int V = OD * OH * OW;
  const int Np = align16i(Co);
  size_t mk = A.off;
  half_t* raw = (half_t*)amalloc(A, (size_t)B * V * Np * sizeof(half_t));
  run_conv(xin, CsIn, Ci, w, raw, B, Co, ID, IH, IW, K, stride, pad,
           OD, OH, OW, A, s);
  run_bn(raw, Np, Co, mask, g, b, nullptr, 0, y, Np, 0, true, B, V, A, s);
  A.off = mk;
}

void run_convt_bn_cat(const half_t* xin, int CsIn, int Ci, const float* w,
                      const float* g, const float* b,
                      const unsigned char* mask, half_t* cat, int CsCat,
                      int B, int Co, int ID, int IH, int IW, Alloc& A,
                      hipStream_t s) {
  const int Vf = 8 * ID * IH * IW;
  const int Np = align16i(Co);
  size_t mk = A.off;
  half_t* raw = (half_t*)amalloc(A, (size_t)B * Vf * Np * sizeof(half_t));
  run_convt(xin, CsIn, Ci, w, raw, B, Co, ID, IH, IW, A, s);
  run_bn(raw, Np, Co, mask, g, b, nullptr, 0, cat, CsCat, 0, true, B, Vf, A, s);
  A.off = mk;
}

void run_copy_cat(const half_t* src, int CsS, int C, half_t* cat, int CsT,
                  int oc0, int V, int B, hipStream_t s) {
  const int total = V * C;
  copy_chan_kernel<<<dim3((total + 255) / 256, B), 256, 0, s>>>(
      src, cat, C, mkmag(C), CsS, CsT, oc0, V, total);
}

void run_block(const BlkP& bk, const half_t* xin, int CsIn, int Ci, int Co,
               int B, int D, int H, int W, const unsigned char* mask,
               half_t* out, Alloc& A, hipStream_t s) {
  const int V = D * H * W;
  const int Np = align16i(Co);
  size_t mk = A.off;
  half_t* raw = (half_t*)amalloc(A, (size_t)B * V * Np * sizeof(half_t));
  run_conv(xin, CsIn, Ci, bk.w1, raw, B, Co, D, H, W, 3, 1, 1, D, H, W, A, s);
  half_t* t1h = (half_t*)amalloc(A, (size_t)B * V * Np * sizeof(half_t));
  run_bn(raw, Np, Co, mask, bk.g1, bk.b1, nullptr, 0, t1h, Np, 0, true,
         B, V, A, s);
  run_conv(t1h, Np, Co, bk.w2, raw /*reuse*/, B, Co, D, H, W, 3, 1, 1,
           D, H, W, A, s);
  const half_t* res = xin;
  int CsRes = CsIn;
  if (bk.wd) {
    half_t* draw = (half_t*)amalloc(A, (size_t)B * V * Np * sizeof(half_t));
    run_conv(xin, CsIn, Ci, bk.wd, draw, B, Co, D, H, W, 1, 1, 0, D, H, W,
             A, s);
    half_t* dh = (half_t*)amalloc(A, (size_t)B * V * Np * sizeof(half_t));
    run_bn(draw, Np, Co, mask, bk.gd, bk.bd, nullptr, 0, dh, Np, 0, false,
           B, V, A, s);
    res = dh; CsRes = Np;
  }
  run_bn(raw, Np, Co, mask, bk.g2, bk.b2, res, CsRes, out, Np, 0, true,
         B, V, A, s);
  A.off = mk;
}

}  // namespace

extern "C" void kernel_launch(void* const* d_in, const int* in_sizes, int n_in,
                              void* d_out, int out_size, void* d_ws,
                              size_t ws_size, hipStream_t stream) {
  (void)in_sizes; (void)n_in; (void)out_size;

  const float* x = (const float*)d_in[0];
  const unsigned char* m1  = (const unsigned char*)d_in[1];
  const unsigned char* m2  = (const unsigned char*)d_in[2];
  const unsigned char* m4  = (const unsigned char*)d_in[3];
  const unsigned char* m8  = (const unsigned char*)d_in[4];
  const unsigned char* m16 = (const unsigned char*)d_in[5];

  int pi = 6;
  auto P = [&]() { return (const float*)d_in[pi++]; };
  auto getBlk = [&](bool hasWd) {
    BlkP k{};
    k.w1 = P(); k.g1 = P(); k.b1 = P();
    k.w2 = P(); k.g2 = P(); k.b2 = P();
    if (hasWd) { k.wd = P(); k.gd = P(); k.bd = P(); }
    return k;
  };

  const float *w0 = P(), *g0 = P(), *b0 = P();
  const float *ws1 = P(), *gs1 = P(), *bs1 = P();
  BlkP blk1 = getBlk(false);
  const float *ws2 = P(), *gs2 = P(), *bs2 = P();
  BlkP blk2 = getBlk(true);
  const float *ws3 = P(), *gs3 = P(), *bs3 = P();
  BlkP blk3 = getBlk(true);
  const float *ws4 = P(), *gs4 = P(), *bs4 = P();
  BlkP blk4 = getBlk(true);
  const float *wt4 = P(), *gt4 = P(), *bt4 = P();
  BlkP blk5 = getBlk(true);
  const float *wt5 = P(), *gt5 = P(), *bt5 = P();
  BlkP blk6 = getBlk(true);
  const float *wt6 = P(), *gt6 = P(), *bt6 = P();
  BlkP blk7 = getBlk(true);
  const float *wt7 = P(), *gt7 = P(), *bt7 = P();
  BlkP blk8 = getBlk(true);
  const float *wf = P(), *bf = P();

  const int B = 2;
  const int D1 = 160, H1 = 160, W1 = 32;
  const int D2 = 80,  H2 = 80,  W2 = 16;
  const int D4 = 40,  H4 = 40,  W4 = 8;
  const int D8 = 20,  H8 = 20,  W8 = 4;
  const int D16 = 10, H16 = 10, W16 = 2;
  const int V1  = D1 * H1 * W1;
  const int V2  = D2 * H2 * W2;
  const int V4  = D4 * H4 * W4;
  const int V8  = D8 * H8 * W8;
  const int V16 = D16 * H16 * W16;

  Alloc A{(char*)d_ws, 0, ws_size};
  hipStream_t s = stream;

  // ---- convert input to f16 channels-last [B][V1][16] (2 real channels) ----
  half_t* xh = (half_t*)amalloc(A, (size_t)B * V1 * 16 * sizeof(half_t));
  {
    const int tot = V1 * 16;
    convert_x_kernel<<<dim3((tot + 255) / 256, B), 256, 0, s>>>(x, xh, V1, tot);
  }

  // ---------------- encoder ----------------
  half_t* p1 = (half_t*)amalloc(A, (size_t)B * V1 * 16 * sizeof(half_t));
  run_conv_bn(xh, 16, 2, w0, g0, b0, m1, p1, B, 8, D1, H1, W1, 5, 1, 2,
              D1, H1, W1, A, s);

  half_t* s1 = (half_t*)amalloc(A, (size_t)B * V2 * 16 * sizeof(half_t));
  run_conv_bn(p1, 16, 8, ws1, gs1, bs1, m2, s1, B, 8, D1, H1, W1, 2, 2, 0,
              D2, H2, W2, A, s);
  half_t* b1 = (half_t*)amalloc(A, (size_t)B * V2 * 16 * sizeof(half_t));
  run_block(blk1, s1, 16, 8, 8, B, D2, H2, W2, m2, b1, A, s);

  half_t* s2 = (half_t*)amalloc(A, (size_t)B * V4 * 16 * sizeof(half_t));
  run_conv_bn(b1, 16, 8, ws2, gs2, bs2, m4, s2, B, 8, D2, H2, W2, 2, 2, 0,
              D4, H4, W4, A, s);
  half_t* b2 = (half_t*)amalloc(A, (size_t)B * V4 * 16 * sizeof(half_t));
  run_block(blk2, s2, 16, 8, 16, B, D4, H4, W4, m4, b2, A, s);

  half_t* s3 = (half_t*)amalloc(A, (size_t)B * V8 * 16 * sizeof(half_t));
  run_conv_bn(b2, 16, 16, ws3, gs3, bs3, m8, s3, B, 16, D4, H4, W4, 2, 2, 0,
              D8, H8, W8, A, s);
  half_t* b3 = (half_t*)amalloc(A, (size_t)B * V8 * 32 * sizeof(half_t));
  run_block(blk3, s3, 16, 16, 32, B, D8, H8, W8, m8, b3, A, s);

  half_t* s4 = (half_t*)amalloc(A, (size_t)B * V16 * 32 * sizeof(half_t));
  run_conv_bn(b3, 32, 32, ws4, gs4, bs4, m16, s4, B, 32, D8, H8, W8, 2, 2, 0,
              D16, H16, W16, A, s);
  half_t* b4 = (half_t*)amalloc(A, (size_t)B * V16 * 64 * sizeof(half_t));
  run_block(blk4, s4, 32, 32, 64, B, D16, H16, W16, m16, b4, A, s);

  // ---------------- decoder ----------------
  half_t* cat5 = (half_t*)amalloc(A, (size_t)B * V8 * 96 * sizeof(half_t));
  (void)hipMemsetAsync(cat5, 0, (size_t)B * V8 * 96 * sizeof(half_t), s);
  run_convt_bn_cat(b4, 64, 64, wt4, gt4, bt4, m8, cat5, 96, B, 64,
                   D16, H16, W16, A, s);
  run_copy_cat(b3, 32, 32, cat5, 96, 64, V8, B, s);
  half_t* b5 = (half_t*)amalloc(A, (size_t)B * V8 * 64 * sizeof(half_t));
  run_block(blk5, cat5, 96, 96, 64, B, D8, H8, W8, m8, b5, A, s);

  half_t* cat6 = (half_t*)amalloc(A, (size_t)B * V4 * 48 * sizeof(half_t));
  (void)hipMemsetAsync(cat6, 0, (size_t)B * V4 * 48 * sizeof(half_t), s);
  run_convt_bn_cat(b5, 64, 64, wt5, gt5, bt5, m4, cat6, 48, B, 32,
                   D8, H8, W8, A, s);
  run_copy_cat(b2, 16, 16, cat6, 48, 32, V4, B, s);
  half_t* b6 = (half_t*)amalloc(A, (size_t)B * V4 * 32 * sizeof(half_t));
  run_block(blk6, cat6, 48, 48, 32, B, D4, H4, W4, m4, b6, A, s);

  half_t* cat7 = (half_t*)amalloc(A, (size_t)B * V2 * 32 * sizeof(half_t));
  (void)hipMemsetAsync(cat7, 0, (size_t)B * V2 * 32 * sizeof(half_t), s);
  run_convt_bn_cat(b6, 32, 32, wt6, gt6, bt6, m2, cat7, 32, B, 16,
                   D4, H4, W4, A, s);
  run_copy_cat(b1, 16, 8, cat7, 32, 16, V2, B, s);
  half_t* b7 = (half_t*)amalloc(A, (size_t)B * V2 * 16 * sizeof(half_t));
  run_block(blk7, cat7, 32, 24, 16, B, D2, H2, W2, m2, b7, A, s);

  half_t* cat8 = (half_t*)amalloc(A, (size_t)B * V1 * 16 * sizeof(half_t));
  (void)hipMemsetAsync(cat8, 0, (size_t)B * V1 * 16 * sizeof(half_t), s);
  run_convt_bn_cat(b7, 16, 16, wt7, gt7, bt7, m1, cat8, 16, B, 8,
                   D2, H2, W2, A, s);
  run_copy_cat(p1, 16, 8, cat8, 16, 8, V1, B, s);
  half_t* b8 = (half_t*)amalloc(A, (size_t)B * V1 * 16 * sizeof(half_t));
  run_block(blk8, cat8, 16, 16, 8, B, D1, H1, W1, m1, b8, A, s);

  // ---------------- head: 1x1 conv + bias + mask + transpose ----------------
  {
    const int HD = H1 * D1;
    final_conv_kernel<<<dim3((HD + 255) / 256, W1, B * 2), 256, 0, s>>>(
        b8, wf, bf, m1, (float*)d_out, D1, mkmag(D1), H1, W1, HD);
  }
}
